// CausalSelfAttention_47880295416535
// MI455X (gfx1250) — matrix operand order repaired and checked by reading, not compiled, not run on hardware
//
#include <hip/hip_runtime.h>
#include <hip/hip_bf16.h>
#include <math.h>

typedef __attribute__((ext_vector_type(16))) _Float16 v16h;
typedef __attribute__((ext_vector_type(8)))  _Float16 h8;
typedef __attribute__((ext_vector_type(4)))  _Float16 h4;
typedef __attribute__((ext_vector_type(8)))  float    v8f;

#define WMMA16(a, b, c) \
    __builtin_amdgcn_wmma_f32_16x16x32_f16(false, (a), false, (b), (short)0, (c), false, false)

static __device__ inline v8f v8zero() {
    v8f z;
#pragma unroll
    for (int i = 0; i < 8; ++i) z[i] = 0.0f;
    return z;
}

// Async global->LDS 16B copy (CDNA5, tracked by ASYNCcnt). LDS operand is the
// wave-relative byte offset == low 32 bits of the generic shared address.
static __device__ inline void async_b128(unsigned lds_off, const _Float16* g) {
    asm volatile("global_load_async_to_lds_b128 %0, %1, off"
                 :: "v"(lds_off), "v"(g) : "memory");
}
static __device__ inline void wait_async0() {
    asm volatile("s_wait_asynccnt 0x0" ::: "memory");
}

// A-fragment 16x32 f16 (ISA 7.12.2): lane l -> row l%16; halves = K {8g..8g+7, 16+8g..23+8g}
static __device__ inline v16h load_a_frag(const _Float16* base, int pitch, int lane) {
    const int row = lane & 15, g = (lane >> 4) & 1;
    const h8* p0 = (const h8*)(base + row * pitch + 8 * g);
    const h8* p1 = (const h8*)(base + row * pitch + 16 + 8 * g);
    h8 lo = *p0, hi = *p1;
    return __builtin_shufflevector(lo, hi, 0, 1, 2, 3, 4, 5, 6, 7, 8, 9, 10, 11, 12, 13, 14, 15);
}

// B-fragment 32x16 f16, LDS stored transposed as [n][k]: lane l -> col l%16, K = {8g..8g+7, 16+8g..23+8g}
static __device__ inline v16h load_b_frag(const _Float16* base, int pitch, int lane) {
    const int n = lane & 15, g = (lane >> 4) & 1;
    const h8* p0 = (const h8*)(base + n * pitch + 8 * g);
    const h8* p1 = (const h8*)(base + n * pitch + 16 + 8 * g);
    h8 lo = *p0, hi = *p1;
    return __builtin_shufflevector(lo, hi, 0, 1, 2, 3, 4, 5, 6, 7, 8, 9, 10, 11, 12, 13, 14, 15);
}

// ---------------------------------------------------------------------------
// Pre-pass: f32 -> f16 convert, and tiled transpose+convert for weights
// ---------------------------------------------------------------------------
__global__ __launch_bounds__(256) void cvt_f16(const float* __restrict__ s,
                                               _Float16* __restrict__ d, int n) {
    const int i = (blockIdx.x * 256 + threadIdx.x) * 4;
    if (i < n) {
        float4 f = *(const float4*)(s + i);
        h4 hv = {(_Float16)f.x, (_Float16)f.y, (_Float16)f.z, (_Float16)f.w};
        *(h4*)(d + i) = hv;
    }
}

// dst[c][r] = (f16) src[r][c];  src is [R][C]
__global__ __launch_bounds__(256) void transpose_f16(const float* __restrict__ src,
                                                     _Float16* __restrict__ dst,
                                                     int R, int C) {
    __shared__ float tile[32][33];
    const int t = threadIdx.x, tx = t & 31, ty = t >> 5;
    const int bx = blockIdx.x * 32, by = blockIdx.y * 32;
#pragma unroll
    for (int i = 0; i < 32; i += 8)
        tile[ty + i][tx] = src[(size_t)(by + ty + i) * C + bx + tx];
    __syncthreads();
#pragma unroll
    for (int i = 0; i < 32; i += 8)
        dst[(size_t)(bx + ty + i) * R + by + tx] = (_Float16)tile[tx][ty + i];
}

// ---------------------------------------------------------------------------
// 128x128-tile GEMM, f16 operands, async double-buffered LDS staging.
// A: f16 [M][K];  Bt: f16 [N][K] (pre-transposed).
// mode 0: scatter to Q/K (f16 [B,H,S,HD], Q * 1/8) and V-transposed [B,H,HD,S]
// mode 1: f32 out + bias
// ---------------------------------------------------------------------------
__global__ __launch_bounds__(256) void gemm128(
    const _Float16* __restrict__ A, const _Float16* __restrict__ Bt,
    const float* __restrict__ bias, float* __restrict__ Cout,
    _Float16* __restrict__ qo, _Float16* __restrict__ ko, _Float16* __restrict__ vTo,
    int M, int N, int K, int mode) {
    __shared__ __align__(16) _Float16 As[2][128 * 40];  // [m][k]
    __shared__ __align__(16) _Float16 Bs[2][128 * 40];  // [n][k]

    const int t = threadIdx.x;
    const int lane = t & 31, w = t >> 5;
    const int g = (lane >> 4) & 1, ln = lane & 15;
    const int Mb = blockIdx.y * 128, Nb = blockIdx.x * 128;
    const int mw = (w >> 1) * 32, nw = (w & 1) * 64;

    v8f c[2][4];
#pragma unroll
    for (int i = 0; i < 2; ++i)
#pragma unroll
        for (int j = 0; j < 4; ++j) c[i][j] = v8zero();

    const int row = t >> 1, kc = (t & 1) * 16;  // each thread stages 32B of A + 32B of B
    auto stage = [&](int k0, int bufi) {
        const _Float16* ga = A + (size_t)(Mb + row) * K + k0 + kc;
        unsigned la = (unsigned)(size_t)&As[bufi][row * 40 + kc];
        async_b128(la, ga);
        async_b128(la + 16, ga + 8);
        const _Float16* gb = Bt + (size_t)(Nb + row) * K + k0 + kc;
        unsigned lb = (unsigned)(size_t)&Bs[bufi][row * 40 + kc];
        async_b128(lb, gb);
        async_b128(lb + 16, gb + 8);
    };

    const int KT = K >> 5;
    stage(0, 0);
    for (int kt = 0; kt < KT; ++kt) {
        wait_async0();
        __syncthreads();
        if (kt + 1 < KT) stage((kt + 1) << 5, (kt + 1) & 1);

        const _Float16* Ab = As[kt & 1];
        const _Float16* Bb = Bs[kt & 1];
        v16h a0 = load_a_frag(&Ab[mw * 40], 40, lane);
        v16h a1 = load_a_frag(&Ab[(mw + 16) * 40], 40, lane);
#pragma unroll
        for (int j = 0; j < 4; ++j) {
            v16h b = load_b_frag(&Bb[(nw + j * 16) * 40], 40, lane);
            c[0][j] = WMMA16(a0, b, c[0][j]);
            c[1][j] = WMMA16(a1, b, c[1][j]);
        }
    }

    // epilogue: C-frag element v -> row v+8g, col ln
#pragma unroll
    for (int i = 0; i < 2; ++i)
#pragma unroll
        for (int j = 0; j < 4; ++j) {
#pragma unroll
            for (int vv = 0; vv < 8; ++vv) {
                const int r = Mb + mw + i * 16 + vv + 8 * g;
                const int cgl = Nb + nw + j * 16 + ln;
                float val = c[i][j][vv] + bias[cgl];
                if (mode == 1) {
                    Cout[(size_t)r * N + cgl] = val;
                } else {
                    const int which = cgl >> 10, rem = cgl & 1023;
                    const int hh = rem >> 6, hd = rem & 63;
                    const int bb = r >> 11, ss = r & 2047;
                    if (which == 0) val *= 0.125f;  // fold 1/sqrt(64) into Q
                    _Float16 hv = (_Float16)val;
                    if (which == 2)
                        vTo[(((size_t)(bb * 16 + hh)) * 64 + hd) * 2048 + ss] = hv;
                    else
                        (which == 0 ? qo : ko)[(((size_t)(bb * 16 + hh)) * 2048 + ss) * 64 + hd] = hv;
                }
            }
        }
}

// ---------------------------------------------------------------------------
// Flash attention: block = 128 queries of one (b,h); 8 waves x 16 query rows.
// K in [B,H,S,HD], V pre-transposed [B,H,HD,S]; async double-buffered staging.
// ---------------------------------------------------------------------------
__global__ __launch_bounds__(256) void attn_flash(
    const _Float16* __restrict__ Qg, const _Float16* __restrict__ Kg,
    const _Float16* __restrict__ Vt, _Float16* __restrict__ ctxh) {
    __shared__ __align__(16) _Float16 qs[128 * 72];      // [q][hd]
    __shared__ __align__(16) _Float16 ks[2][32 * 72];    // [key][hd] (B-layout for QK^T)
    __shared__ __align__(16) _Float16 vts[2][64 * 40];   // [hd][key] (B-layout for P@V)
    __shared__ __align__(16) _Float16 ps[8 * 16 * 40];   // per-wave P tile [row][key]

    const int t = threadIdx.x;
    const int lane = t & 31, w = t >> 5;
    const int g = (lane >> 4) & 1, ln = lane & 15;
    const int qb = blockIdx.x;  // query tile 0..15
    const int bh = blockIdx.y;  // b*16+h
    const size_t head = (size_t)bh * 2048 * 64;  // also valid for Vt ([64][2048] per head)
    const int q0 = qb * 128 + w * 16;

    {  // stage Q tile 128x64 via async copy
        const int row = t >> 1, cb = (t & 1) * 32;
        const _Float16* gq = Qg + head + (size_t)(qb * 128 + row) * 64 + cb;
        unsigned lq = (unsigned)(size_t)&qs[row * 72 + cb];
        async_b128(lq, gq);
        async_b128(lq + 16, gq + 8);
        async_b128(lq + 32, gq + 16);
        async_b128(lq + 48, gq + 24);
    }
    wait_async0();
    __syncthreads();
    const v16h qa0 = load_a_frag(&qs[w * 16 * 72], 72, lane);       // hd 0..31
    const v16h qa1 = load_a_frag(&qs[w * 16 * 72 + 32], 72, lane);  // hd 32..63

    auto stage_kv = [&](int j, int bufi) {
        const int key = t >> 3, cb = (t & 7) * 8;
        async_b128((unsigned)(size_t)&ks[bufi][key * 72 + cb],
                   Kg + head + (size_t)(j * 32 + key) * 64 + cb);
        const int hd = t >> 2, kc = (t & 3) * 8;
        async_b128((unsigned)(size_t)&vts[bufi][hd * 40 + kc],
                   Vt + head + (size_t)hd * 2048 + j * 32 + kc);
    };

    float m[8], ll[8];
    v8f o[4];
#pragma unroll
    for (int vv = 0; vv < 8; ++vv) { m[vv] = -INFINITY; ll[vv] = 0.0f; }
#pragma unroll
    for (int n = 0; n < 4; ++n) o[n] = v8zero();

    const int jend = qb * 4 + 3;
    stage_kv(0, 0);
    for (int j = 0; j <= jend; ++j) {
        wait_async0();
        __syncthreads();
        if (j < jend) stage_kv(j + 1, (j + 1) & 1);

        if (j * 32 <= q0 + 15) {
            const _Float16* kb = ks[j & 1];
            const _Float16* vb = vts[j & 1];
            v8f s0 = v8zero(), s1 = v8zero();
            s0 = WMMA16(qa0, load_b_frag(&kb[0], 72, lane), s0);
            s0 = WMMA16(qa1, load_b_frag(&kb[32], 72, lane), s0);
            s1 = WMMA16(qa0, load_b_frag(&kb[16 * 72], 72, lane), s1);
            s1 = WMMA16(qa1, load_b_frag(&kb[16 * 72 + 32], 72, lane), s1);

            const int kc0 = j * 32 + ln, kc1 = kc0 + 16;
            float alpha[8];
#pragma unroll
            for (int vv = 0; vv < 8; ++vv) {
                const int qr = q0 + vv + 8 * g;
                if (kc0 > qr) s0[vv] = -INFINITY;
                if (kc1 > qr) s1[vv] = -INFINITY;
                float mt = fmaxf(s0[vv], s1[vv]);
#pragma unroll
                for (int off = 1; off < 16; off <<= 1) mt = fmaxf(mt, __shfl_xor(mt, off, 16));
                const float mn = fmaxf(m[vv], mt);
                alpha[vv] = expf(m[vv] - mn);
                const float p0 = expf(s0[vv] - mn);
                const float p1 = expf(s1[vv] - mn);
                s0[vv] = p0; s1[vv] = p1;
                float rs = p0 + p1;
#pragma unroll
                for (int off = 1; off < 16; off <<= 1) rs += __shfl_xor(rs, off, 16);
                ll[vv] = ll[vv] * alpha[vv] + rs;
                m[vv] = mn;
            }
#pragma unroll
            for (int n = 0; n < 4; ++n)
#pragma unroll
                for (int vv = 0; vv < 8; ++vv) o[n][vv] *= alpha[vv];

            // P -> per-wave LDS -> A-fragment layout (same-wave DS ops are in-order)
            _Float16* pw = &ps[w * 16 * 40];
#pragma unroll
            for (int vv = 0; vv < 8; ++vv) {
                pw[(vv + 8 * g) * 40 + ln] = (_Float16)s0[vv];
                pw[(vv + 8 * g) * 40 + 16 + ln] = (_Float16)s1[vv];
            }
            const v16h pa = load_a_frag(pw, 40, lane);
#pragma unroll
            for (int n = 0; n < 4; ++n) {
                v16h vbf = load_b_frag(&vb[n * 16 * 40], 40, lane);
                o[n] = WMMA16(pa, vbf, o[n]);
            }
        }
    }

    // normalize + write f16 context in [b*2048+s][h*64+hd] layout
    const int bb = bh >> 4, hh = bh & 15;
#pragma unroll
    for (int n = 0; n < 4; ++n)
#pragma unroll
        for (int vv = 0; vv < 8; ++vv) {
            const int r = q0 + vv + 8 * g;
            const int col = hh * 64 + n * 16 + ln;
            ctxh[(size_t)(bb * 2048 + r) * 1024 + col] = (_Float16)(o[n][vv] / ll[vv]);
        }
}

extern "C" void kernel_launch(void* const* d_in, const int* in_sizes, int n_in,
                              void* d_out, int out_size, void* d_ws, size_t ws_size,
                              hipStream_t stream) {
    (void)in_sizes; (void)n_in; (void)out_size; (void)ws_size;
    const float* x      = (const float*)d_in[0];
    const float* w_qkv  = (const float*)d_in[1];
    const float* b_qkv  = (const float*)d_in[2];
    const float* w_proj = (const float*)d_in[3];
    const float* b_proj = (const float*)d_in[4];
    float* out = (float*)d_out;

    char* ws = (char*)d_ws;
    const size_t MB = 1024 * 1024;
    _Float16* xh     = (_Float16*)(ws);              // 8 MiB  [4096][1024]
    _Float16* wqkvT  = (_Float16*)(ws + 8 * MB);     // 6 MiB  [3072][1024]
    _Float16* wprojT = (_Float16*)(ws + 14 * MB);    // 2 MiB  [1024][1024]
    _Float16* q      = (_Float16*)(ws + 16 * MB);    // 8 MiB  [B,H,S,HD]
    _Float16* k      = (_Float16*)(ws + 24 * MB);    // 8 MiB  [B,H,S,HD]
    _Float16* vT     = (_Float16*)(ws + 32 * MB);    // 8 MiB  [B,H,HD,S]
    _Float16* ctxh   = (_Float16*)(ws + 40 * MB);    // 8 MiB  [4096][1024]

    // 0) pre-pass: convert x; transpose+convert weights (so hot loops are pure f16 copies)
    cvt_f16<<<(4096 * 1024) / 1024, 256, 0, stream>>>(x, xh, 4096 * 1024);
    transpose_f16<<<dim3(3072 / 32, 1024 / 32), 256, 0, stream>>>(w_qkv, wqkvT, 1024, 3072);
    transpose_f16<<<dim3(1024 / 32, 1024 / 32), 256, 0, stream>>>(w_proj, wprojT, 1024, 1024);

    // 1) QKV projection -> Q,K f16 [B,H,S,HD]; V f16 transposed [B,H,HD,S]
    gemm128<<<dim3(3072 / 128, 4096 / 128), 256, 0, stream>>>(
        xh, wqkvT, b_qkv, nullptr, q, k, vT, 4096, 3072, 1024, 0);

    // 2) causal flash attention -> ctx f16 [4096][1024]
    attn_flash<<<dim3(16, 32), 256, 0, stream>>>(q, k, vT, ctxh);

    // 3) output projection -> f32 out + bias
    gemm128<<<dim3(1024 / 128, 4096 / 128), 256, 0, stream>>>(
        ctxh, wprojT, b_proj, out, nullptr, nullptr, nullptr, 4096, 1024, 1024, 1);
}
